// BloomAttention_21251498180849
// MI455X (gfx1250) — matrix operand order repaired and checked by reading, not compiled, not run on hardware
//
#include <hip/hip_runtime.h>
#include <hip/hip_bf16.h>
#include <stdint.h>

// Problem constants (B=1)
#define S_LEN 2048
#define HID   4096
#define NH    32
#define HD    128
#define K3    (3 * HID) // 12288
#define INV_NORM 0.08838834764831845f // 1/sqrt(128)

typedef __attribute__((ext_vector_type(16))) __bf16         v16bf;
typedef __attribute__((ext_vector_type(8)))  float          v8f;
typedef __attribute__((ext_vector_type(16))) unsigned short v16u;
typedef __attribute__((ext_vector_type(8)))  unsigned short v8u;
typedef __attribute__((ext_vector_type(4)))  unsigned int   v4u;
typedef __attribute__((ext_vector_type(8)))  int            v8i;
typedef __attribute__((ext_vector_type(4)))  int            v4i;

// Toolchain discriminator for the TDM builtin arity (see CDNA5_HIP.md):
// the gfx1250 TDM header ships only with the 6-arg (therock/clang-23) toolchain.
#if __has_include(<hip/amd_detail/amd_gfx1250_TDM.h>)
#define TDM_6ARG 1
#else
#define TDM_6ARG 0
#endif

// float -> bf16 bits, round-to-nearest-even
__device__ __forceinline__ unsigned short f2bf(float f) {
    unsigned int x = __float_as_uint(f);
    x += 0x7FFFu + ((x >> 16) & 1u);
    return (unsigned short)(x >> 16);
}

__device__ __forceinline__ v16u combine8(v8u lo, v8u hi) {
    return __builtin_shufflevector(lo, hi, 0,1,2,3,4,5,6,7,8,9,10,11,12,13,14,15);
}

// D = A(16x32 bf16) x B(32x16 bf16) + C(16x16 f32)
__device__ __forceinline__ v8f wmma_bf16(v16u a, v16u b, v8f c) {
    v16bf ab = __builtin_bit_cast(v16bf, a);
    v16bf bb = __builtin_bit_cast(v16bf, b);
    return __builtin_amdgcn_wmma_f32_16x16x32_bf16(
        false, ab, false, bb, (short)0, c, false, false);
}

// 16-bit A-matrix 16x32 fragment: contiguous K, half hh holds k = 16*hh .. 16*hh+15.
__device__ __forceinline__ v16u load_afrag(const unsigned short* aPtr, int kk, int hh) {
    v8u lo = *(const v8u*)(aPtr + kk + hh * 16);
    v8u hi = *(const v8u*)(aPtr + kk + hh * 16 + 8);
    return combine8(lo, hi);
}

// ---------------------------------------------------------------------------
// TDM: async 2D tile DMA global -> LDS (D# per CDNA5 ISA)
// ---------------------------------------------------------------------------
__device__ __forceinline__ void tdm_load_2d(unsigned lds_addr, const void* gptr,
                                            unsigned dim0, unsigned dim1,
                                            unsigned stride0,
                                            unsigned tile0, unsigned tile1) {
    unsigned long long ga = (unsigned long long)gptr;
    v4u g0;
    g0[0] = 1u;                                   // count=1, user descriptor
    g0[1] = lds_addr;                             // lds_addr [63:32]
    g0[2] = (unsigned)(ga & 0xFFFFFFFFu);         // global_addr [95:64]
    g0[3] = (unsigned)((ga >> 32) & 0x01FFFFFFu)  // global_addr [120:96]
          | (2u << 30);                           // type=2 ("image") [127:126]
    unsigned w[8];
    w[0] = (1u << 16);                                            // data_size=2B
    w[1] = (dim0 & 0xFFFFu) << 16;                                // tensor_dim0 lo
    w[2] = ((dim0 >> 16) & 0xFFFFu) | ((dim1 & 0xFFFFu) << 16);   // dim0 hi | dim1 lo
    w[3] = ((dim1 >> 16) & 0xFFFFu) | ((tile0 & 0xFFFFu) << 16);  // dim1 hi | tile_dim0
    w[4] = (tile1 & 0xFFFFu);                                     // tile_dim1
    w[5] = stride0;                                               // dim0_stride lo32
    w[6] = 0u;
    w[7] = 0u;
    v8i g1;
#pragma unroll
    for (int i = 0; i < 8; ++i) g1[i] = (int)w[i];
    v4i z4 = {0, 0, 0, 0};
#if TDM_6ARG
    v8i z8 = {0, 0, 0, 0, 0, 0, 0, 0};
    __builtin_amdgcn_tensor_load_to_lds(g0, g1, z4, z4, z8, 0);
#else
    __builtin_amdgcn_tensor_load_to_lds(g0, g1, z4, z4, 0);
#endif
}

// Generic address of a __shared__ object: low 32 bits are the LDS byte address.
#define LDS_OFF(p) ((unsigned)(unsigned long long)(const void*)(p))

// ---------------------------------------------------------------------------
// 1) fp32 -> bf16 conversion (grid-stride)
// ---------------------------------------------------------------------------
__global__ void cvt_f32_bf16_kernel(const float* __restrict__ in,
                                    unsigned short* __restrict__ out, int n) {
    int i = blockIdx.x * blockDim.x + threadIdx.x;
    int stride = gridDim.x * blockDim.x;
    for (; i < n; i += stride) out[i] = f2bf(in[i]);
}

// ---------------------------------------------------------------------------
// 2) QKV GEMM: fused = X[2048,4096] @ Wqkv^T + b.
//    Wave tile 32x64 (acc[2][4]); ping-pong register pipeline (no rotation
//    copies). Epilogue scatters into Q (pre-scaled), K, V-transposed.
// ---------------------------------------------------------------------------
__global__ __launch_bounds__(256)
void qkv_gemm_kernel(const unsigned short* __restrict__ X,   // [S_LEN][HID] bf16
                     const unsigned short* __restrict__ W,   // [K3][HID]   bf16
                     const float* __restrict__ bias,         // [K3]
                     unsigned short* __restrict__ Qg,        // [NH][S][HD]
                     unsigned short* __restrict__ Kg,        // [NH][S][HD]
                     unsigned short* __restrict__ Vt)        // [NH][HD][S]
{
    const int lane = threadIdx.x & 31;
    const int wave = threadIdx.x >> 5;
    const int wm = wave & 1;
    const int wn = wave >> 1;
    const int mBase = blockIdx.y * 64 + wm * 32;   // 2 x 16-row tiles per wave
    const int nBase = blockIdx.x * 256 + wn * 64;
    const int lr = lane & 15;
    const int hh = lane >> 4;

    v8f acc[2][4] = {};
    const unsigned short* aPtr0 = X + (mBase + lr) * HID;
    const unsigned short* aPtr1 = X + (mBase + 16 + lr) * HID;
    const unsigned short* bPtr[4];
#pragma unroll
    for (int nb = 0; nb < 4; ++nb)
        bPtr[nb] = W + (nBase + nb * 16 + lr) * HID + hh * 16;

    v16u aA[2], bA[4], aB[2], bB[4];
    aA[0] = load_afrag(aPtr0, 0, hh);
    aA[1] = load_afrag(aPtr1, 0, hh);
#pragma unroll
    for (int nb = 0; nb < 4; ++nb) bA[nb] = *(const v16u*)(bPtr[nb]);

    int kk = 0;
    for (; kk < HID - 64; kk += 64) {
        __builtin_prefetch(aPtr0 + kk + 256, 0, 3);
        __builtin_prefetch(aPtr1 + kk + 256, 0, 3);
        // Half-step 1: load set B (kk+32), compute with set A (kk).
        aB[0] = load_afrag(aPtr0, kk + 32, hh);
        aB[1] = load_afrag(aPtr1, kk + 32, hh);
#pragma unroll
        for (int nb = 0; nb < 4; ++nb)
            bB[nb] = *(const v16u*)(bPtr[nb] + kk + 32);
#pragma unroll
        for (int nb = 0; nb < 4; ++nb) {
            acc[0][nb] = wmma_bf16(aA[0], bA[nb], acc[0][nb]);
            acc[1][nb] = wmma_bf16(aA[1], bA[nb], acc[1][nb]);
        }
        // Half-step 2: load set A (kk+64), compute with set B (kk+32).
        aA[0] = load_afrag(aPtr0, kk + 64, hh);
        aA[1] = load_afrag(aPtr1, kk + 64, hh);
#pragma unroll
        for (int nb = 0; nb < 4; ++nb)
            bA[nb] = *(const v16u*)(bPtr[nb] + kk + 64);
#pragma unroll
        for (int nb = 0; nb < 4; ++nb) {
            acc[0][nb] = wmma_bf16(aB[0], bB[nb], acc[0][nb]);
            acc[1][nb] = wmma_bf16(aB[1], bB[nb], acc[1][nb]);
        }
    }
    // Tail: set A holds chunk kk; final chunk kk+32 == HID-32.
    aB[0] = load_afrag(aPtr0, kk + 32, hh);
    aB[1] = load_afrag(aPtr1, kk + 32, hh);
#pragma unroll
    for (int nb = 0; nb < 4; ++nb)
        bB[nb] = *(const v16u*)(bPtr[nb] + kk + 32);
#pragma unroll
    for (int nb = 0; nb < 4; ++nb) {
        acc[0][nb] = wmma_bf16(aA[0], bA[nb], acc[0][nb]);
        acc[1][nb] = wmma_bf16(aA[1], bA[nb], acc[1][nb]);
    }
#pragma unroll
    for (int nb = 0; nb < 4; ++nb) {
        acc[0][nb] = wmma_bf16(aB[0], bB[nb], acc[0][nb]);
        acc[1][nb] = wmma_bf16(aB[1], bB[nb], acc[1][nb]);
    }

#pragma unroll
    for (int nb = 0; nb < 4; ++nb) {
        const int col = nBase + nb * 16 + lr;
        const int h   = col / (3 * HD);
        const int rem = col % (3 * HD);
        const int t   = rem / HD;
        const int d   = rem % HD;
        const float bv = bias[col];
#pragma unroll
        for (int wm2 = 0; wm2 < 2; ++wm2) {
#pragma unroll
            for (int r = 0; r < 8; ++r) {
                const int srow = mBase + wm2 * 16 + r + hh * 8;
                const float v = acc[wm2][nb][r] + bv;
                if (t == 0)      Qg[(h * S_LEN + srow) * HD + d] = f2bf(v * INV_NORM);
                else if (t == 1) Kg[(h * S_LEN + srow) * HD + d] = f2bf(v);
                else             Vt[(h * HD + d) * S_LEN + srow] = f2bf(v);
            }
        }
    }
}

// ---------------------------------------------------------------------------
// 3) Flash attention with ALiBi + causal mask.
//    One wave per (head, 16-row query tile); 32-key steps; K/V tiles streamed
//    into double-buffered LDS by the Tensor Data Mover.
// ---------------------------------------------------------------------------
__global__ __launch_bounds__(32)
void attn_kernel(const unsigned short* __restrict__ Qg,   // [NH][S][HD] pre-scaled
                 const unsigned short* __restrict__ Kg,   // [NH][S][HD]
                 const unsigned short* __restrict__ Vt,   // [NH][HD][S]
                 const float* __restrict__ alibi,         // [NH][S]
                 unsigned short* __restrict__ ctx)        // [S][HID] bf16
{
    __shared__ unsigned short Kls[2][32][128];  // [buf][keyLocal][d]
    __shared__ unsigned short Vls[2][128][32];  // [buf][d][keyLocal]
    __shared__ unsigned short Pls[16][32];      // probs tile

    const int lane = threadIdx.x;
    const int lr = lane & 15;
    const int hh = lane >> 4;
    const int mBase = blockIdx.x * 16;
    const int h = blockIdx.y;

    v16u aq[4];
    const unsigned short* qPtr = Qg + (h * S_LEN + mBase + lr) * HD;
#pragma unroll
    for (int dc = 0; dc < 4; ++dc) aq[dc] = load_afrag(qPtr, dc * 32, hh);

    v8f o[8] = {};
    float mrow[8], lrow[8];
#pragma unroll
    for (int r = 0; r < 8; ++r) { mrow[r] = -3.0e38f; lrow[r] = 0.0f; }

    const int nsteps = (mBase + 16 + 31) / 32;

    auto issue_tiles = [&](int st2, int b2) {
        tdm_load_2d(LDS_OFF(&Kls[b2][0][0]),
                    Kg + ((size_t)h * S_LEN + (size_t)st2 * 32) * HD,
                    HD, S_LEN, HD, /*tile0=*/HD, /*tile1=*/32);
        tdm_load_2d(LDS_OFF(&Vls[b2][0][0]),
                    Vt + (size_t)h * HD * S_LEN + (size_t)st2 * 32,
                    S_LEN, HD, S_LEN, /*tile0=*/32, /*tile1=*/HD);
    };

    issue_tiles(0, 0);

    for (int st = 0; st < nsteps; ++st) {
        const int keyBase = st * 32;
        const int buf = st & 1;
        if (st + 1 < nsteps) {
            issue_tiles(st + 1, buf ^ 1);
            __builtin_amdgcn_s_wait_tensorcnt((short)2);
        } else {
            __builtin_amdgcn_s_wait_tensorcnt((short)0);
        }
        asm volatile("" ::: "memory"); // keep LDS reads below the wait

        v8f s0 = {}, s1 = {};
#pragma unroll
        for (int dc = 0; dc < 4; ++dc) {
            v16u b0 = *(const v16u*)(&Kls[buf][lr][dc * 32 + hh * 16]);
            v16u b1 = *(const v16u*)(&Kls[buf][16 + lr][dc * 32 + hh * 16]);
            s0 = wmma_bf16(aq[dc], b0, s0);
            s1 = wmma_bf16(aq[dc], b1, s1);
        }

        const float al0 = alibi[h * S_LEN + keyBase + lr];
        const float al1 = alibi[h * S_LEN + keyBase + 16 + lr];
        const int key0 = keyBase + lr;
        const int key1 = keyBase + 16 + lr;
#pragma unroll
        for (int r = 0; r < 8; ++r) {
            const int M = r + hh * 8;
            const int grow = mBase + M;
            float v0 = s0[r] + al0;
            float v1 = s1[r] + al1;
            if (key0 > grow) v0 = -3.0e38f;
            if (key1 > grow) v1 = -3.0e38f;
            float cand = fmaxf(v0, v1);
#pragma unroll
            for (int off = 8; off >= 1; off >>= 1)
                cand = fmaxf(cand, __shfl_xor(cand, off, 32));
            const float mnew  = fmaxf(mrow[r], cand);
            const float scale = __expf(mrow[r] - mnew);
            const float p0 = __expf(v0 - mnew);
            const float p1 = __expf(v1 - mnew);
            float ps = p0 + p1;
#pragma unroll
            for (int off = 8; off >= 1; off >>= 1)
                ps += __shfl_xor(ps, off, 32);
            lrow[r] = lrow[r] * scale + ps;
            mrow[r] = mnew;
#pragma unroll
            for (int ch = 0; ch < 8; ++ch) o[ch][r] *= scale;
            Pls[M][lr]      = f2bf(p0);
            Pls[M][16 + lr] = f2bf(p1);
        }
        __syncthreads();

        v8u plo = *(const v8u*)(&Pls[lr][hh * 16]);
        v8u phi = *(const v8u*)(&Pls[lr][hh * 16 + 8]);
        v16u ap = combine8(plo, phi);
#pragma unroll
        for (int ch = 0; ch < 8; ++ch) {
            v16u bv = *(const v16u*)(&Vls[buf][ch * 16 + lr][hh * 16]);
            o[ch] = wmma_bf16(ap, bv, o[ch]);
        }
        __syncthreads();
    }

#pragma unroll
    for (int r = 0; r < 8; ++r) {
        const int M = r + hh * 8;
        const float inv = 1.0f / lrow[r];
#pragma unroll
        for (int ch = 0; ch < 8; ++ch)
            ctx[(mBase + M) * HID + h * HD + ch * 16 + lr] = f2bf(o[ch][r] * inv);
    }
}

// ---------------------------------------------------------------------------
// 4) Dense GEMM: out = ctx @ Wd^T + b_dense (fp32 out); same ping-pong scheme.
// ---------------------------------------------------------------------------
__global__ __launch_bounds__(256)
void dense_gemm_kernel(const unsigned short* __restrict__ A,  // [S_LEN][HID] bf16
                       const unsigned short* __restrict__ W,  // [HID][HID]   bf16
                       const float* __restrict__ bias,        // [HID]
                       float* __restrict__ out)               // [S_LEN][HID]
{
    const int lane = threadIdx.x & 31;
    const int wave = threadIdx.x >> 5;
    const int wm = wave & 1;
    const int wn = wave >> 1;
    const int mBase = blockIdx.y * 64 + wm * 32;
    const int nBase = blockIdx.x * 256 + wn * 64;
    const int lr = lane & 15;
    const int hh = lane >> 4;

    v8f acc[2][4] = {};
    const unsigned short* aPtr0 = A + (mBase + lr) * HID;
    const unsigned short* aPtr1 = A + (mBase + 16 + lr) * HID;
    const unsigned short* bPtr[4];
#pragma unroll
    for (int nb = 0; nb < 4; ++nb)
        bPtr[nb] = W + (nBase + nb * 16 + lr) * HID + hh * 16;

    v16u aA[2], bA[4], aB[2], bB[4];
    aA[0] = load_afrag(aPtr0, 0, hh);
    aA[1] = load_afrag(aPtr1, 0, hh);
#pragma unroll
    for (int nb = 0; nb < 4; ++nb) bA[nb] = *(const v16u*)(bPtr[nb]);

    int kk = 0;
    for (; kk < HID - 64; kk += 64) {
        __builtin_prefetch(aPtr0 + kk + 256, 0, 3);
        __builtin_prefetch(aPtr1 + kk + 256, 0, 3);
        aB[0] = load_afrag(aPtr0, kk + 32, hh);
        aB[1] = load_afrag(aPtr1, kk + 32, hh);
#pragma unroll
        for (int nb = 0; nb < 4; ++nb)
            bB[nb] = *(const v16u*)(bPtr[nb] + kk + 32);
#pragma unroll
        for (int nb = 0; nb < 4; ++nb) {
            acc[0][nb] = wmma_bf16(aA[0], bA[nb], acc[0][nb]);
            acc[1][nb] = wmma_bf16(aA[1], bA[nb], acc[1][nb]);
        }
        aA[0] = load_afrag(aPtr0, kk + 64, hh);
        aA[1] = load_afrag(aPtr1, kk + 64, hh);
#pragma unroll
        for (int nb = 0; nb < 4; ++nb)
            bA[nb] = *(const v16u*)(bPtr[nb] + kk + 64);
#pragma unroll
        for (int nb = 0; nb < 4; ++nb) {
            acc[0][nb] = wmma_bf16(aB[0], bB[nb], acc[0][nb]);
            acc[1][nb] = wmma_bf16(aB[1], bB[nb], acc[1][nb]);
        }
    }
    aB[0] = load_afrag(aPtr0, kk + 32, hh);
    aB[1] = load_afrag(aPtr1, kk + 32, hh);
#pragma unroll
    for (int nb = 0; nb < 4; ++nb)
        bB[nb] = *(const v16u*)(bPtr[nb] + kk + 32);
#pragma unroll
    for (int nb = 0; nb < 4; ++nb) {
        acc[0][nb] = wmma_bf16(aA[0], bA[nb], acc[0][nb]);
        acc[1][nb] = wmma_bf16(aA[1], bA[nb], acc[1][nb]);
    }
#pragma unroll
    for (int nb = 0; nb < 4; ++nb) {
        acc[0][nb] = wmma_bf16(aB[0], bB[nb], acc[0][nb]);
        acc[1][nb] = wmma_bf16(aB[1], bB[nb], acc[1][nb]);
    }

#pragma unroll
    for (int nb = 0; nb < 4; ++nb) {
        const int col = nBase + nb * 16 + lr;
        const float bv = bias[col];
#pragma unroll
        for (int wm2 = 0; wm2 < 2; ++wm2) {
#pragma unroll
            for (int r = 0; r < 8; ++r) {
                const int srow = mBase + wm2 * 16 + r + hh * 8;
                out[srow * HID + col] = acc[wm2][nb][r] + bv;
            }
        }
    }
}

// ---------------------------------------------------------------------------
// Launcher
// ---------------------------------------------------------------------------
extern "C" void kernel_launch(void* const* d_in, const int* in_sizes, int n_in,
                              void* d_out, int out_size, void* d_ws, size_t ws_size,
                              hipStream_t stream) {
    (void)in_sizes; (void)n_in; (void)out_size; (void)ws_size;

    const float* hidden  = (const float*)d_in[0]; // [1,2048,4096]
    const float* alibi   = (const float*)d_in[1]; // [32,1,2048]
    const float* w_qkv   = (const float*)d_in[2]; // [12288,4096]
    const float* b_qkv   = (const float*)d_in[3]; // [12288]
    const float* w_dense = (const float*)d_in[4]; // [4096,4096]
    const float* b_dense = (const float*)d_in[5]; // [4096]
    float* out = (float*)d_out;                   // [2048,4096]

    char* ws = (char*)d_ws;
    size_t off = 0;
    auto alloc_bytes = [&](size_t bytes) -> void* {
        void* p = ws + off;
        off = (off + bytes + 255) & ~size_t(255);
        return p;
    };
    unsigned short* Xb    = (unsigned short*)alloc_bytes((size_t)S_LEN * HID * 2);
    unsigned short* Wqkvb = (unsigned short*)alloc_bytes((size_t)K3 * HID * 2);
    unsigned short* Wdb   = (unsigned short*)alloc_bytes((size_t)HID * HID * 2);
    unsigned short* Qb    = (unsigned short*)alloc_bytes((size_t)NH * S_LEN * HD * 2);
    unsigned short* Kb    = (unsigned short*)alloc_bytes((size_t)NH * S_LEN * HD * 2);
    unsigned short* Vtb   = (unsigned short*)alloc_bytes((size_t)NH * HD * S_LEN * 2);
    unsigned short* Ctxb  = (unsigned short*)alloc_bytes((size_t)S_LEN * HID * 2);

    cvt_f32_bf16_kernel<<<dim3(2048), 256, 0, stream>>>(hidden, Xb, S_LEN * HID);
    cvt_f32_bf16_kernel<<<dim3(4096), 256, 0, stream>>>(w_qkv, Wqkvb, K3 * HID);
    cvt_f32_bf16_kernel<<<dim3(2048), 256, 0, stream>>>(w_dense, Wdb, HID * HID);

    qkv_gemm_kernel<<<dim3(K3 / 256, S_LEN / 64), 256, 0, stream>>>(
        Xb, Wqkvb, b_qkv, Qb, Kb, Vtb);

    attn_kernel<<<dim3(S_LEN / 16, NH), 32, 0, stream>>>(Qb, Kb, Vtb, alibi, Ctxb);

    dense_gemm_kernel<<<dim3(HID / 256, S_LEN / 64), 256, 0, stream>>>(
        Ctxb, Wdb, b_dense, out);
}
